// WeightOnlyInt8Linear_55585466745293
// MI455X (gfx1250) — matrix operand order repaired and checked by reading, not compiled, not run on hardware
//
#include <hip/hip_runtime.h>
#include <hip/hip_bf16.h>

typedef __attribute__((ext_vector_type(16))) _Float16 f16x16;
typedef __attribute__((ext_vector_type(8)))  float    f32x8;

#define IN_F     4096
#define OUT_F    11008
#define NGROUPS  32

#define BM 128
#define BN 128
#define BK 64
#define LDSK 72   // BK + 8 halves pad -> row stride 144B = 36 dwords (conflict-free b128 frags)
#define KTILES (IN_F / BK)   // 64

__global__ __launch_bounds__(256) void wq8_linear_wmma(
    const float* __restrict__ A,   // [4096][4096] fp32 activations
    const int*   __restrict__ W,   // [11008][4096] int8 values in int32
    const float* __restrict__ S,   // [11008][32] group scales
    float* __restrict__ C)         // [4096][11008]
{
    __shared__ __align__(16) _Float16 sA[2][BM][LDSK];
    __shared__ __align__(16) _Float16 sB[2][BN][LDSK];

    const int tid   = threadIdx.x;
    const int lane  = tid & 31;
    const int wid   = tid >> 5;
    const int wm    = (wid & 3) * 32;   // wave M offset inside tile
    const int wn    = (wid >> 2) * 64;  // wave N offset inside tile
    const int l16   = lane & 15;
    const int khalf = lane >> 4;        // 0: lanes 0-15, 1: lanes 16-31

    const int m0 = blockIdx.x * BM;     // x sweeps M fastest: weight tile stays hot in L2
    const int n0 = blockIdx.y * BN;

    // staging registers (double-buffer prefetch)
    float4 rA[8];
    int4   rW[8];
    float  rS[8];

    auto load_tile = [&](int k0) {
#pragma unroll
        for (int i = 0; i < 8; ++i) {
            const int f   = tid + i * 256;      // float4 index, 0..2047
            const int row = f >> 4;             // 16 float4 per row
            const int col = (f & 15) << 2;
            rA[i] = *(const float4*)(A + (size_t)(m0 + row) * IN_F + k0 + col);
            rW[i] = *(const int4*)(W + (size_t)(n0 + row) * IN_F + k0 + col);
            rS[i] = S[(n0 + row) * NGROUPS + (k0 >> 7)];
        }
    };
    auto store_tile = [&](int buf) {
#pragma unroll
        for (int i = 0; i < 8; ++i) {
            const int f   = tid + i * 256;
            const int row = f >> 4;
            const int col = (f & 15) << 2;
            {   // activations: fp32 -> f16
                union { _Float16 h[4]; uint2 u; } cv;
                cv.h[0] = (_Float16)rA[i].x; cv.h[1] = (_Float16)rA[i].y;
                cv.h[2] = (_Float16)rA[i].z; cv.h[3] = (_Float16)rA[i].w;
                *(uint2*)&sA[buf][row][col] = cv.u;
            }
            {   // weights: int8-in-int32 * group scale -> f16 (int8 exact in f16)
                union { _Float16 h[4]; uint2 u; } cv;
                const float s = rS[i];
                cv.h[0] = (_Float16)((float)rW[i].x * s);
                cv.h[1] = (_Float16)((float)rW[i].y * s);
                cv.h[2] = (_Float16)((float)rW[i].z * s);
                cv.h[3] = (_Float16)((float)rW[i].w * s);
                *(uint2*)&sB[buf][row][col] = cv.u;
            }
        }
    };

    f32x8 acc[2][4];
#pragma unroll
    for (int i = 0; i < 2; ++i)
#pragma unroll
        for (int j = 0; j < 4; ++j)
#pragma unroll
            for (int e = 0; e < 8; ++e) acc[i][j][e] = 0.0f;

    // prologue: stage tile 0
    load_tile(0);
    store_tile(0);
    __syncthreads();

    for (int t = 0; t < KTILES; ++t) {
        const int cur = t & 1;

        // issue next tile's global loads now; their waits land after the WMMA burst
        if (t + 1 < KTILES) load_tile((t + 1) * BK);

        // ---- compute from current buffer: 2 x (2 A-frags, 4 B-frags, 8 WMMAs)
#pragma unroll
        for (int kk = 0; kk < BK; kk += 32) {
            f16x16 afrag[2];
#pragma unroll
            for (int mi = 0; mi < 2; ++mi) {
                // A 16x32 f16 layout: lane<16 -> K 0-7 & 16-23; lane>=16 -> K 8-15 & 24-31
                const _Float16* p = &sA[cur][wm + mi * 16 + l16][kk + khalf * 8];
                union { uint4 q[2]; f16x16 v; } u;
                u.q[0] = *(const uint4*)p;
                u.q[1] = *(const uint4*)(p + 16);
                afrag[mi] = u.v;
            }
#pragma unroll
            for (int ni = 0; ni < 4; ++ni) {
                // B 32x16 f16 layout: lane<16 -> K 0-7 & 16-23; lane>=16 -> K 8-15 & 24-31 (col = lane%16)
                const _Float16* p = &sB[cur][wn + ni * 16 + l16][kk + khalf * 8];
                union { uint4 q[2]; f16x16 v; } u;
                u.q[0] = *(const uint4*)p;
                u.q[1] = *(const uint4*)(p + 16);
                const f16x16 bfrag = u.v;
                acc[0][ni] = __builtin_amdgcn_wmma_f32_16x16x32_f16(
                    false, afrag[0], false, bfrag, (short)0, acc[0][ni], false, false);
                acc[1][ni] = __builtin_amdgcn_wmma_f32_16x16x32_f16(
                    false, afrag[1], false, bfrag, (short)0, acc[1][ni], false, false);
            }
        }

        // convert + stash next tile into the alternate buffer, then one barrier
        if (t + 1 < KTILES) store_tile((t + 1) & 1);
        __syncthreads();
    }

    // ---- writeback: C/D 16x16 f32 layout -> VGPR r holds row (r + 8*khalf), col = lane%16
    const int mBase = m0 + wm + khalf * 8;
#pragma unroll
    for (int mi = 0; mi < 2; ++mi) {
#pragma unroll
        for (int ni = 0; ni < 4; ++ni) {
            const int n = n0 + wn + ni * 16 + l16;
#pragma unroll
            for (int r = 0; r < 8; ++r) {
                C[(size_t)(mBase + mi * 16 + r) * OUT_F + n] = acc[mi][ni][r];
            }
        }
    }
}

extern "C" void kernel_launch(void* const* d_in, const int* in_sizes, int n_in,
                              void* d_out, int out_size, void* d_ws, size_t ws_size,
                              hipStream_t stream) {
    (void)in_sizes; (void)n_in; (void)out_size; (void)d_ws; (void)ws_size;
    const float* A = (const float*)d_in[0];
    const int*   W = (const int*)d_in[1];
    const float* S = (const float*)d_in[2];
    float*       C = (float*)d_out;

    dim3 grid(4096 / BM, OUT_F / BN);   // (32, 86); x (M) varies fastest
    wq8_linear_wmma<<<grid, 256, 0, stream>>>(A, W, S, C);
}
